// GConvGRU_73555609911720
// MI455X (gfx1250) — matrix operand order repaired and checked by reading, not compiled, not run on hardware
//
#include <hip/hip_runtime.h>
#include <math.h>

// ---------------------------------------------------------------------------
// GConvGRU (ChebConv K=3 + GRU) for MI455X / gfx1250.
//   * SpMM propagation: wave-per-edge, float2 per lane, global f32 atomics
//     (memory-bound part, ~20 GB total -> ~1 ms at 23.3 TB/s).
//   * Dense Chebyshev GEMMs: v_wmma_f32_16x16x32_bf16 with bf16x3 split
//     precision (A_hi*B_hi + A_hi*B_lo + A_lo*B_hi), f32 accumulation,
//     native v_cvt_pk_bf16_f32 conversion.
//   * Z and R gates fused into one kernel (shared A fragments -> half the
//     A-side traffic/converts); T2 = 2*P2 - T0 folded into fragment build;
//     bias + activation fused into the GEMM epilogue.
//   * sched_barrier per N-tile inside the accumulate loop bounds in-flight
//     B fragments to one tile (32 VGPRs) -> live set ~170 VGPRs, no spills.
// ---------------------------------------------------------------------------

#define N_NODES 50000
#define N_EDGES 800000
#define CH      64
#define SEQ     8
#define NC      (N_NODES * CH)      // 3,200,000 floats per [N,64] buffer

typedef __attribute__((ext_vector_type(16))) __bf16 v16bf;
typedef __attribute__((ext_vector_type(8)))  float  v8f;

// Stop the scheduler from hoisting the next tile's loads over this tile's
// WMMAs (bounds register pressure; each tile still overlaps 8 b128 loads
// with 6 WMMAs, and multi-wave occupancy hides the rest).
__device__ __forceinline__ void sched_fence() {
#if defined(__has_builtin)
#if __has_builtin(__builtin_amdgcn_sched_barrier)
  __builtin_amdgcn_sched_barrier(0);
#endif
#endif
}

// ---------------- bf16 helpers (native HW converts, RNE) ----------------
__device__ __forceinline__ void split_bf16(float v, __bf16* hi, __bf16* lo) {
  __bf16 h = (__bf16)v;            // native f32->bf16 convert (RNE)
  *hi = h;
  *lo = (__bf16)(v - (float)h);    // residual for split-precision product
}

// ---------------- small utility kernels ----------------
__global__ void fill_zero_kernel(float* __restrict__ p, int n4) {
  int i = blockIdx.x * blockDim.x + threadIdx.x;
  if (i < n4) ((float4*)p)[i] = make_float4(0.f, 0.f, 0.f, 0.f);
}

__global__ void deg_kernel(const int* __restrict__ row, const int* __restrict__ col,
                           const float* __restrict__ ew, float* __restrict__ deg, int E) {
  int e = blockIdx.x * blockDim.x + threadIdx.x;
  if (e >= E) return;
  int r = row[e];
  float w = (r == col[e]) ? 0.f : ew[e];     // PyG removes self loops
  atomicAdd(&deg[r], w);
}

__global__ void dinv_kernel(const float* __restrict__ deg, float* __restrict__ dinv, int n) {
  int i = blockIdx.x * blockDim.x + threadIdx.x;
  if (i >= n) return;
  float d = deg[i];
  dinv[i] = (d > 0.f) ? rsqrtf(d) : 0.f;
}

// L_hat off-diagonal entry: -(2/lambda_max) * w * dinv[row]*dinv[col];
// lambda_max = 2 -> factor 1; diagonal entry is exactly 0.
__global__ void wn_kernel(const int* __restrict__ row, const int* __restrict__ col,
                          const float* __restrict__ ew, const float* __restrict__ dinv,
                          float* __restrict__ wn, int E) {
  int e = blockIdx.x * blockDim.x + threadIdx.x;
  if (e >= E) return;
  int r = row[e], c = col[e];
  float w = (r == c) ? 0.f : ew[e];
  wn[e] = -w * dinv[r] * dinv[c];
}

// out[row] += wn * v[col]  (one wave32 per edge, float2 per lane == 64 ch)
__global__ __launch_bounds__(256) void spmm_kernel(
    const int* __restrict__ row, const int* __restrict__ col,
    const float* __restrict__ wn, const float* __restrict__ v,
    float* __restrict__ out, int E) {
  int gid  = blockIdx.x * blockDim.x + threadIdx.x;
  int e    = gid >> 5;
  int lane = gid & 31;
  if (e >= E) return;
  float w = wn[e];
  int c = col[e], r = row[e];
  float2 vv = ((const float2*)(v + (size_t)c * CH))[lane];
  float* o = out + (size_t)r * CH + lane * 2;
  atomicAdd(o,     w * vv.x);
  atomicAdd(o + 1, w * vv.y);
}

// Transpose W[k][cin][hid] f32 -> WT_hi/WT_lo[k][hid][cin] bf16 (split precision)
__global__ void wt_kernel(const float* __restrict__ w0, const float* __restrict__ w1,
                          const float* __restrict__ w2, const float* __restrict__ w3,
                          const float* __restrict__ w4, const float* __restrict__ w5,
                          unsigned short* __restrict__ WTh, unsigned short* __restrict__ WTl) {
  int i = blockIdx.x * blockDim.x + threadIdx.x;   // over 6*3*64*64
  if (i >= 6 * 3 * 4096) return;
  int m   = i / (3 * 4096);
  int rem = i % (3 * 4096);
  int k   = rem / 4096;
  int nc  = rem % 4096;
  int n = nc / 64, c = nc % 64;
  const float* W = (m == 0) ? w0 : (m == 1) ? w1 : (m == 2) ? w2
                  : (m == 3) ? w3 : (m == 4) ? w4 : w5;
  float val = W[k * 4096 + c * 64 + n];
  __bf16 hi, lo;
  split_bf16(val, &hi, &lo);
  WTh[i] = __builtin_bit_cast(unsigned short, hi);
  WTl[i] = __builtin_bit_cast(unsigned short, lo);
}

// ---------------- WMMA fragment loaders ----------------
// A 16x32 bf16: lane m = l&15; lanes 0-15 hold K {0..15},
// lanes 16-31 hold K {16..31}; each 8-K run is one contiguous 32B load.

// Plain load + hi/lo split (Chebyshev T0 and T1 terms).
__device__ __forceinline__ void make_afrag_one(
    const float* __restrict__ A, int row0, int lane, int ks, v16bf* hi, v16bf* lo) {
  int m   = lane & 15;
  int off = ks * 32 + ((lane & 16) ? 16 : 0);
  const float* pa = A + (size_t)(row0 + m) * CH + off;
  v16bf h, l;
#pragma unroll
  for (int half = 0; half < 2; ++half) {
    float4 x0 = ((const float4*)(pa + half * 8))[0];
    float4 x1 = ((const float4*)(pa + half * 8))[1];
    float vals[8] = { x0.x, x0.y, x0.z, x0.w, x1.x, x1.y, x1.z, x1.w };
#pragma unroll
    for (int j = 0; j < 8; ++j) {
      __bf16 hb, lb;
      split_bf16(vals[j], &hb, &lb);
      h[half * 8 + j] = hb;
      l[half * 8 + j] = lb;
    }
  }
  *hi = h; *lo = l;
}

// T2 = 2*P2 - T0 folded at load time (Chebyshev recurrence term).
__device__ __forceinline__ void make_afrag_t2(
    const float* __restrict__ T0p, const float* __restrict__ P2p,
    int row0, int lane, int ks, v16bf* hi, v16bf* lo) {
  int m   = lane & 15;
  int off = ks * 32 + ((lane & 16) ? 16 : 0);
  const float* pa = T0p + (size_t)(row0 + m) * CH + off;
  const float* pb = P2p + (size_t)(row0 + m) * CH + off;
  v16bf h, l;
#pragma unroll
  for (int half = 0; half < 2; ++half) {
    float4 xa0 = ((const float4*)(pa + half * 8))[0];
    float4 xa1 = ((const float4*)(pa + half * 8))[1];
    float4 xb0 = ((const float4*)(pb + half * 8))[0];
    float4 xb1 = ((const float4*)(pb + half * 8))[1];
    float vals[8] = { 2.f*xb0.x - xa0.x, 2.f*xb0.y - xa0.y,
                      2.f*xb0.z - xa0.z, 2.f*xb0.w - xa0.w,
                      2.f*xb1.x - xa1.x, 2.f*xb1.y - xa1.y,
                      2.f*xb1.z - xa1.z, 2.f*xb1.w - xa1.w };
#pragma unroll
    for (int j = 0; j < 8; ++j) {
      __bf16 hb, lb;
      split_bf16(vals[j], &hb, &lb);
      h[half * 8 + j] = hb;
      l[half * 8 + j] = lb;
    }
  }
  *hi = h; *lo = l;
}

// B 32x16 bf16: N = nt*16 + (l&15); K = ks*32 + (l>=16 ? 16 : 0) + j,
// j=0..15 contiguous in pre-transposed WT[hid][cin] -> two uint4 loads.
__device__ __forceinline__ v16bf load_bfrag(const unsigned short* __restrict__ WT,
                                            int lane, int nt, int ks) {
  int n  = nt * 16 + (lane & 15);
  int kb = ks * 32 + ((lane & 16) ? 16 : 0);
  const uint4* p = (const uint4*)(WT + n * 64 + kb);
  union { v16bf v; uint4 q[2]; } r;
  r.q[0] = p[0]; r.q[1] = p[1];
  return r.v;
}

// Accumulate one Chebyshev term (one A matrix x one 64x64 weight) into 4
// N-tiles with bf16x3 split precision: Ah*Bh + Al*Bh + Ah*Bl.
// A sched_fence per N-tile keeps at most one tile's B fragments in flight.
__device__ __forceinline__ void accum_term(
    v8f acc[4], v16bf a0h, v16bf a0l, v16bf a1h, v16bf a1l,
    const unsigned short* __restrict__ wth, const unsigned short* __restrict__ wtl,
    int lane) {
#pragma unroll
  for (int nt = 0; nt < 4; ++nt) {
    v16bf bh0 = load_bfrag(wth, lane, nt, 0);
    v16bf bl0 = load_bfrag(wtl, lane, nt, 0);
    acc[nt] = __builtin_amdgcn_wmma_f32_16x16x32_bf16(false, a0h, false, bh0, (short)0, acc[nt], false, false);
    acc[nt] = __builtin_amdgcn_wmma_f32_16x16x32_bf16(false, a0l, false, bh0, (short)0, acc[nt], false, false);
    acc[nt] = __builtin_amdgcn_wmma_f32_16x16x32_bf16(false, a0h, false, bl0, (short)0, acc[nt], false, false);
    v16bf bh1 = load_bfrag(wth, lane, nt, 1);
    v16bf bl1 = load_bfrag(wtl, lane, nt, 1);
    acc[nt] = __builtin_amdgcn_wmma_f32_16x16x32_bf16(false, a1h, false, bh1, (short)0, acc[nt], false, false);
    acc[nt] = __builtin_amdgcn_wmma_f32_16x16x32_bf16(false, a1l, false, bh1, (short)0, acc[nt], false, false);
    acc[nt] = __builtin_amdgcn_wmma_f32_16x16x32_bf16(false, a1h, false, bl1, (short)0, acc[nt], false, false);
    sched_fence();
  }
}

// Store one gate tile: bias + activation + scatter per D-matrix layout
// (VGPR r -> row = r + 8*(lane>=16), col = lane&15 per N-tile).
template <int ACT>   // 0 = sigmoid, 1 = tanh
__device__ __forceinline__ void store_gate(
    const v8f acc[4], const float* __restrict__ bx, const float* __restrict__ bh,
    float* __restrict__ out, int row0, int lane) {
  int rb = row0 + ((lane & 16) ? 8 : 0);
#pragma unroll
  for (int nt = 0; nt < 4; ++nt) {
    int colg = nt * 16 + (lane & 15);
    float bias = bx[colg] + bh[colg];
#pragma unroll
    for (int r = 0; r < 8; ++r) {
      float v = acc[nt][r] + bias;
      if (ACT == 0) v = 1.0f / (1.0f + __expf(-v));
      else          v = tanhf(v);
      out[(size_t)(rb + r) * CH + colg] = v;
    }
  }
}

// ---------------- fused Z+R gate kernel ----------------
// Z = sigmoid(cheb(X;Wxz)+cheb(H;Whz)), R = sigmoid(cheb(X;Wxr)+cheb(H;Whr)).
// Both gates share the same six A matrices -> build fragments once.
__global__ __launch_bounds__(256) void cheb_zr_kernel(
    const float* __restrict__ A0x, const float* __restrict__ P1x, const float* __restrict__ P2x,
    const float* __restrict__ A0h, const float* __restrict__ P1h, const float* __restrict__ P2h,
    const unsigned short* __restrict__ Wzxh, const unsigned short* __restrict__ Wzxl,
    const unsigned short* __restrict__ Wzhh, const unsigned short* __restrict__ Wzhl,
    const unsigned short* __restrict__ Wrxh, const unsigned short* __restrict__ Wrxl,
    const unsigned short* __restrict__ Wrhh, const unsigned short* __restrict__ Wrhl,
    const float* __restrict__ bzx, const float* __restrict__ bzh,
    const float* __restrict__ brx, const float* __restrict__ brh,
    float* __restrict__ Zout, float* __restrict__ Rout) {
  int wave = blockIdx.x * (blockDim.x >> 5) + (threadIdx.x >> 5);
  int row0 = wave * 16;
  if (row0 >= N_NODES) return;              // whole-wave uniform exit
  int lane = threadIdx.x & 31;

  v8f accZ[4], accR[4];
#pragma unroll
  for (int nt = 0; nt < 4; ++nt) { accZ[nt] = (v8f){0,0,0,0,0,0,0,0};
                                   accR[nt] = (v8f){0,0,0,0,0,0,0,0}; }

  const float* T0[2] = {A0x, A0h};
  const float* P1[2] = {P1x, P1h};
  const float* P2[2] = {P2x, P2h};
  const unsigned short* WZH[2] = {Wzxh, Wzhh};
  const unsigned short* WZL[2] = {Wzxl, Wzhl};
  const unsigned short* WRH[2] = {Wrxh, Wrhh};
  const unsigned short* WRL[2] = {Wrxl, Wrhl};

#pragma unroll
  for (int s = 0; s < 2; ++s) {
    v16bf a0h, a0l, a1h, a1l;
    // c = 0 : T0
    make_afrag_one(T0[s], row0, lane, 0, &a0h, &a0l);
    make_afrag_one(T0[s], row0, lane, 1, &a1h, &a1l);
    accum_term(accZ, a0h, a0l, a1h, a1l, WZH[s] + 0*4096, WZL[s] + 0*4096, lane);
    accum_term(accR, a0h, a0l, a1h, a1l, WRH[s] + 0*4096, WRL[s] + 0*4096, lane);
    // c = 1 : T1 = P1
    make_afrag_one(P1[s], row0, lane, 0, &a0h, &a0l);
    make_afrag_one(P1[s], row0, lane, 1, &a1h, &a1l);
    accum_term(accZ, a0h, a0l, a1h, a1l, WZH[s] + 1*4096, WZL[s] + 1*4096, lane);
    accum_term(accR, a0h, a0l, a1h, a1l, WRH[s] + 1*4096, WRL[s] + 1*4096, lane);
    // c = 2 : T2 = 2*P2 - T0
    make_afrag_t2(T0[s], P2[s], row0, lane, 0, &a0h, &a0l);
    make_afrag_t2(T0[s], P2[s], row0, lane, 1, &a1h, &a1l);
    accum_term(accZ, a0h, a0l, a1h, a1l, WZH[s] + 2*4096, WZL[s] + 2*4096, lane);
    accum_term(accR, a0h, a0l, a1h, a1l, WRH[s] + 2*4096, WRL[s] + 2*4096, lane);
  }

  store_gate<0>(accZ, bzx, bzh, Zout, row0, lane);
  store_gate<0>(accR, brx, brh, Rout, row0, lane);
}

// ---------------- single-gate kernel (Ht, tanh) ----------------
__global__ __launch_bounds__(256) void cheb_ht_kernel(
    const float* __restrict__ A0x, const float* __restrict__ P1x, const float* __restrict__ P2x,
    const float* __restrict__ A0h, const float* __restrict__ P1h, const float* __restrict__ P2h,
    const unsigned short* __restrict__ WTxh, const unsigned short* __restrict__ WTxl,
    const unsigned short* __restrict__ WThh, const unsigned short* __restrict__ WThl,
    const float* __restrict__ bx, const float* __restrict__ bh,
    float* __restrict__ out) {
  int wave = blockIdx.x * (blockDim.x >> 5) + (threadIdx.x >> 5);
  int row0 = wave * 16;
  if (row0 >= N_NODES) return;
  int lane = threadIdx.x & 31;

  v8f acc[4];
#pragma unroll
  for (int nt = 0; nt < 4; ++nt) acc[nt] = (v8f){0,0,0,0,0,0,0,0};

  const float* T0[2] = {A0x, A0h};
  const float* P1[2] = {P1x, P1h};
  const float* P2[2] = {P2x, P2h};
  const unsigned short* WH[2] = {WTxh, WThh};
  const unsigned short* WL[2] = {WTxl, WThl};

#pragma unroll
  for (int s = 0; s < 2; ++s) {
    v16bf a0h, a0l, a1h, a1l;
    make_afrag_one(T0[s], row0, lane, 0, &a0h, &a0l);
    make_afrag_one(T0[s], row0, lane, 1, &a1h, &a1l);
    accum_term(acc, a0h, a0l, a1h, a1l, WH[s] + 0*4096, WL[s] + 0*4096, lane);
    make_afrag_one(P1[s], row0, lane, 0, &a0h, &a0l);
    make_afrag_one(P1[s], row0, lane, 1, &a1h, &a1l);
    accum_term(acc, a0h, a0l, a1h, a1l, WH[s] + 1*4096, WL[s] + 1*4096, lane);
    make_afrag_t2(T0[s], P2[s], row0, lane, 0, &a0h, &a0l);
    make_afrag_t2(T0[s], P2[s], row0, lane, 1, &a1h, &a1l);
    accum_term(acc, a0h, a0l, a1h, a1l, WH[s] + 2*4096, WL[s] + 2*4096, lane);
  }

  store_gate<1>(acc, bx, bh, out, row0, lane);
}

// ---------------- elementwise GRU kernels ----------------
__global__ void hr_kernel(const float* __restrict__ H, const float* __restrict__ R,
                          float* __restrict__ HR, int n) {
  int i = blockIdx.x * blockDim.x + threadIdx.x;
  if (i < n) HR[i] = H[i] * R[i];
}

__global__ void update_kernel(const float* __restrict__ H, const float* __restrict__ Z,
                              const float* __restrict__ Ht, float* __restrict__ Hnew,
                              float* __restrict__ states, float* __restrict__ hlast, int n) {
  int i = blockIdx.x * blockDim.x + threadIdx.x;
  if (i >= n) return;
  float z  = Z[i];
  float hn = z * H[i] + (1.f - z) * Ht[i];
  Hnew[i]   = hn;
  states[i] = hn;
  if (hlast) hlast[i] = hn;
}

// ---------------------------------------------------------------------------
extern "C" void kernel_launch(void* const* d_in, const int* in_sizes, int n_in,
                              void* d_out, int out_size, void* d_ws, size_t ws_size,
                              hipStream_t stream) {
  const float* X  = (const float*)d_in[0];     // [1,8,N,64]
  const int*   ei = (const int*)d_in[1];       // [2,E]
  const float* ew = (const float*)d_in[2];     // [E]
  // mats: 0 xz, 1 hz, 2 xr, 3 hr, 4 xh, 5 hh
  const float* W[6] = {(const float*)d_in[3],  (const float*)d_in[5],  (const float*)d_in[7],
                       (const float*)d_in[9],  (const float*)d_in[11], (const float*)d_in[13]};
  const float* B[6] = {(const float*)d_in[4],  (const float*)d_in[6],  (const float*)d_in[8],
                       (const float*)d_in[10], (const float*)d_in[12], (const float*)d_in[14]};
  const int* row = ei;
  const int* col = ei + N_EDGES;

  float* ws   = (float*)d_ws;
  float* Hb   = ws + (size_t)0 * NC;
  float* Zb   = ws + (size_t)1 * NC;
  float* Rb   = ws + (size_t)2 * NC;   // reused for Ht after HR is formed
  float* HRb  = ws + (size_t)3 * NC;
  float* PX1  = ws + (size_t)4 * NC;
  float* PX2  = ws + (size_t)5 * NC;
  float* PA1  = ws + (size_t)6 * NC;   // PH1 then PR1
  float* PA2  = ws + (size_t)7 * NC;   // PH2 then PR2
  float* deg  = ws + (size_t)8 * NC;
  float* dinv = deg + N_NODES;
  float* wn   = dinv + N_NODES;
  unsigned short* WTh = (unsigned short*)(wn + N_EDGES);   // [6][3][64][64] bf16 hi
  unsigned short* WTl = WTh + 6 * 3 * 4096;                // [6][3][64][64] bf16 lo

  float* states = (float*)d_out;        // [SEQ][N][64]
  float* hlast  = states + (size_t)SEQ * NC;

  const int T256 = 256;
  int g_fill = (NC / 4 + T256 - 1) / T256;               // float4 fill
  int g_edge = (N_EDGES + T256 - 1) / T256;
  int g_node = (N_NODES + T256 - 1) / T256;
  int g_spmm = (N_EDGES * 32 + T256 - 1) / T256;
  int g_wt   = (6 * 3 * 4096 + T256 - 1) / T256;
  int g_ew   = (NC + T256 - 1) / T256;
  int g_gemm = ((N_NODES / 16) + 7) / 8;                 // 8 waves / block

  // --- edge normalization (diag term is exactly 0 for lambda_max=2) ---
  fill_zero_kernel<<<(N_NODES/4 + T256-1)/T256, T256, 0, stream>>>(deg, N_NODES/4);
  deg_kernel <<<g_edge, T256, 0, stream>>>(row, col, ew, deg, N_EDGES);
  dinv_kernel<<<g_node, T256, 0, stream>>>(deg, dinv, N_NODES);
  wn_kernel  <<<g_edge, T256, 0, stream>>>(row, col, ew, dinv, wn, N_EDGES);

  // --- weights: transpose + split into bf16 hi/lo ---
  wt_kernel<<<g_wt, T256, 0, stream>>>(W[0], W[1], W[2], W[3], W[4], W[5], WTh, WTl);

  // --- H0 = 0 ---
  fill_zero_kernel<<<g_fill, T256, 0, stream>>>(Hb, NC / 4);

  const int MAT = 3 * 4096;   // ushorts per weight tensor
  for (int t = 0; t < SEQ; ++t) {
    const float* Xt = X + (size_t)t * NC;

    // Chebyshev propagations shared across gates
    fill_zero_kernel<<<g_fill, T256, 0, stream>>>(PX1, NC / 4);
    spmm_kernel<<<g_spmm, T256, 0, stream>>>(row, col, wn, Xt, PX1, N_EDGES);
    fill_zero_kernel<<<g_fill, T256, 0, stream>>>(PX2, NC / 4);
    spmm_kernel<<<g_spmm, T256, 0, stream>>>(row, col, wn, PX1, PX2, N_EDGES);
    fill_zero_kernel<<<g_fill, T256, 0, stream>>>(PA1, NC / 4);
    spmm_kernel<<<g_spmm, T256, 0, stream>>>(row, col, wn, Hb, PA1, N_EDGES);
    fill_zero_kernel<<<g_fill, T256, 0, stream>>>(PA2, NC / 4);
    spmm_kernel<<<g_spmm, T256, 0, stream>>>(row, col, wn, PA1, PA2, N_EDGES);

    // Fused Z + R gates (shared A fragments)
    cheb_zr_kernel<<<g_gemm, T256, 0, stream>>>(
        Xt, PX1, PX2, Hb, PA1, PA2,
        WTh + 0*MAT, WTl + 0*MAT, WTh + 1*MAT, WTl + 1*MAT,   // Wxz, Whz
        WTh + 2*MAT, WTl + 2*MAT, WTh + 3*MAT, WTl + 3*MAT,   // Wxr, Whr
        B[0], B[1], B[2], B[3], Zb, Rb);

    // HR = H * R, then its propagations (reuse PA1/PA2)
    hr_kernel<<<g_ew, T256, 0, stream>>>(Hb, Rb, HRb, NC);
    fill_zero_kernel<<<g_fill, T256, 0, stream>>>(PA1, NC / 4);
    spmm_kernel<<<g_spmm, T256, 0, stream>>>(row, col, wn, HRb, PA1, N_EDGES);
    fill_zero_kernel<<<g_fill, T256, 0, stream>>>(PA2, NC / 4);
    spmm_kernel<<<g_spmm, T256, 0, stream>>>(row, col, wn, PA1, PA2, N_EDGES);

    // Ht = tanh(cheb(X;W_xh) + cheb(H*R;W_hh))  (writes over Rb)
    cheb_ht_kernel<<<g_gemm, T256, 0, stream>>>(
        Xt, PX1, PX2, HRb, PA1, PA2,
        WTh + 4*MAT, WTl + 4*MAT, WTh + 5*MAT, WTl + 5*MAT, B[4], B[5], Rb);

    // H = Z*H + (1-Z)*Ht ; emit state slice (+ Hlast on final step)
    update_kernel<<<g_ew, T256, 0, stream>>>(
        Hb, Zb, Rb, Hb, states + (size_t)t * NC,
        (t == SEQ - 1) ? hlast : (float*)nullptr, NC);
  }
}
